// TransformerMoEEncoderLayer_49108656063221
// MI455X (gfx1250) — matrix operand order repaired and checked by reading, compiled, not run on hardware
//
#include <hip/hip_runtime.h>
#include <hip/hip_bf16.h>

// ============================================================================
// TransformerMoEEncoderLayer for MI455X (gfx1250, wave32, WMMA).
// Memory-bound (expert weights = 537 MB/pass -> ~28 us HBM floor at 23.3 TB/s).
// GEMMs: v_wmma_f32_16x16x32_f16 with double-buffered
// GLOBAL_LOAD_ASYNC_TO_LDS_B128 tile staging (ASYNCcnt ping-pong).
// Softmax / layernorm / routing stay in fp32 VALU.
// ============================================================================

typedef __attribute__((ext_vector_type(16))) _Float16 v16h;
typedef __attribute__((ext_vector_type(8)))  float    v8f;

namespace {
constexpr int S_ = 2048, B_ = 8, D_ = 512, H_ = 8, F_ = 2048, E_ = 64;
constexpr int N_ = S_ * B_;          // 16384 tokens
constexpr int CAP_ = 2 * N_ / E_;    // 512 capacity per expert
}

static __device__ inline v8f v8fz() {
  v8f z = {0.f, 0.f, 0.f, 0.f, 0.f, 0.f, 0.f, 0.f};
  return z;
}

// Generic pointer to a __shared__ object: low 32 bits are the LDS byte address
// (flat aperture scheme, ISA 10.2).
static __device__ inline unsigned lds_a32(const void* p) {
  return (unsigned)(unsigned long long)p;
}

// Async DMA 16 bytes global -> LDS per lane (no VGPR data, ASYNCcnt-tracked).
static __device__ inline void async_cp16(unsigned lds, const void* g) {
  asm volatile("global_load_async_to_lds_b128 %0, %1, off"
               :: "v"(lds), "v"(g) : "memory");
}

// K index inside 16x32 f16 A fragment for element e (0..15), lane-group g.
// (contiguous order matching B fragments: lanes 0-15 hold K 0..15; lanes 16-31 hold K 16..31)
static __device__ inline int a_kidx(int e, int g) {
  return e + 16 * g;
}

// ---- f16-LDS fragment gathers (attention) ----------------------------------
static __device__ inline v16h frag_a(const _Float16* p, int stride) {
  int lane = threadIdx.x & 31, g = lane >> 4, m = lane & 15;
  v16h a;
#pragma unroll
  for (int e = 0; e < 16; ++e) a[e] = p[m * stride + a_kidx(e, g)];
  return a;
}
static __device__ inline v16h frag_b(const _Float16* p, int stride) {
  int lane = threadIdx.x & 31, g = lane >> 4, n = lane & 15;
  v16h b;
#pragma unroll
  for (int e = 0; e < 16; ++e) b[e] = p[(e + 16 * g) * stride + n];
  return b;
}
static __device__ inline v16h frag_bt(const _Float16* p, int stride) {
  int lane = threadIdx.x & 31, g = lane >> 4, n = lane & 15;
  v16h b;
#pragma unroll
  for (int e = 0; e < 16; ++e) b[e] = p[n * stride + e + 16 * g];
  return b;
}

// ---- fp32-LDS fragment gathers with on-the-fly cvt (GEMM path) -------------
static __device__ inline v16h frag_a_f32(const float* p, int stride) {
  int lane = threadIdx.x & 31, g = lane >> 4, m = lane & 15;
  v16h a;
#pragma unroll
  for (int e = 0; e < 16; ++e) a[e] = (_Float16)p[m * stride + a_kidx(e, g)];
  return a;
}
static __device__ inline v16h frag_b_f32(const float* p, int stride) { // (k,n)=p[k*st+n]
  int lane = threadIdx.x & 31, g = lane >> 4, n = lane & 15;
  v16h b;
#pragma unroll
  for (int e = 0; e < 16; ++e) b[e] = (_Float16)p[(e + 16 * g) * stride + n];
  return b;
}
static __device__ inline v16h frag_bt_f32(const float* p, int stride) { // (k,n)=p[n*st+k]
  int lane = threadIdx.x & 31, g = lane >> 4, n = lane & 15;
  v16h b;
#pragma unroll
  for (int e = 0; e < 16; ++e) b[e] = (_Float16)p[n * stride + e + 16 * g];
  return b;
}

// ============================================================================
// Generic tiled GEMM:  C[z] = op( A[z] (M x K)  *  B[z]  + bias[z] )
//   TB=true : B is (Nc x K) row-major, used transposed (C = A * B^T)
//   TB=false: B is (K x Nc) row-major
// 64x64 tile / 256 threads (8 waves, each 16 rows x 32 cols).
// Tiles staged fp32 via async-LDS DMA, double-buffered (4 async ops/wave/tile,
// in-order completion -> s_wait_asynccnt 4 retires exactly the current tile).
// ============================================================================
template <bool TB, bool RELU>
__global__ __launch_bounds__(256)
void gemm_f16_wmma(const float* __restrict__ A, int lda, long aSz,
                   const float* __restrict__ Bm, int ldb, long bSz,
                   const float* __restrict__ bias, long biasSz,
                   float* __restrict__ C, int ldc, long cSz,
                   int K, int acc)
{
  __shared__ float Al[2][64 * 32];   // [row][k]
  __shared__ float Bl[2][2048];      // TB: [n][k] (64x32), !TB: [k][n] (32x64)
  const int z = blockIdx.z;
  A  += (long)z * aSz;
  Bm += (long)z * bSz;
  C  += (long)z * cSz;
  const int n0 = blockIdx.x * 64;
  const int m0 = blockIdx.y * 64;
  const int t = threadIdx.x;
  const int wave = t >> 5, wm = wave & 3, wn = wave >> 2;
  v8f acc0 = v8fz(), acc1 = v8fz();

  auto issue = [&](int buf, int k0) {
    // A tile 64x32 fp32: 512 x 16B transfers, 2 per thread
#pragma unroll
    for (int j = 0; j < 2; ++j) {
      int x = t + j * 256;
      int row = x >> 3, c4 = (x & 7) << 2;
      async_cp16(lds_a32(&Al[buf][row * 32 + c4]),
                 &A[(long)(m0 + row) * lda + k0 + c4]);
    }
    if (TB) {        // global contiguous along k -> LDS [n][k]
#pragma unroll
      for (int j = 0; j < 2; ++j) {
        int x = t + j * 256;
        int n = x >> 3, c4 = (x & 7) << 2;
        async_cp16(lds_a32(&Bl[buf][n * 32 + c4]),
                   &Bm[(long)(n0 + n) * ldb + k0 + c4]);
      }
    } else {         // global contiguous along n -> LDS [k][n]
#pragma unroll
      for (int j = 0; j < 2; ++j) {
        int x = t + j * 256;
        int kk = x >> 4, c4 = (x & 15) << 2;
        async_cp16(lds_a32(&Bl[buf][kk * 64 + c4]),
                   &Bm[(long)(k0 + kk) * ldb + n0 + c4]);
      }
    }
  };

  issue(0, 0);
  const int nch = K >> 5;
  for (int ci = 0; ci < nch; ++ci) {
    const int buf = ci & 1;
    if (ci + 1 < nch) {
      issue(buf ^ 1, (ci + 1) << 5);                 // prefetch next tile
      asm volatile("s_wait_asynccnt 0x4" ::: "memory");  // current tile landed
    } else {
      asm volatile("s_wait_asynccnt 0x0" ::: "memory");
    }
    __syncthreads();   // make all waves' async tile pieces visible
    v16h a = frag_a_f32(&Al[buf][wm * 16 * 32], 32);
    v16h b0, b1;
    if (TB) {
      b0 = frag_bt_f32(&Bl[buf][(wn * 32) * 32], 32);
      b1 = frag_bt_f32(&Bl[buf][(wn * 32 + 16) * 32], 32);
    } else {
      b0 = frag_b_f32(&Bl[buf][wn * 32], 64);
      b1 = frag_b_f32(&Bl[buf][wn * 32 + 16], 64);
    }
    acc0 = __builtin_amdgcn_wmma_f32_16x16x32_f16(false, a, false, b0,
                                                  (short)0, acc0, false, false);
    acc1 = __builtin_amdgcn_wmma_f32_16x16x32_f16(false, a, false, b1,
                                                  (short)0, acc1, false, false);
    __syncthreads();   // done reading buf before it is overwritten
  }

  const int lane = t & 31, g = lane >> 4, nn = lane & 15;
  const int col0 = n0 + wn * 32 + nn;
  float bA = 0.f, bB = 0.f;
  if (bias) {
    const float* bp = bias + (long)z * biasSz;
    bA = bp[col0];
    bB = bp[col0 + 16];
  }
#pragma unroll
  for (int r = 0; r < 8; ++r) {
    const int row = m0 + wm * 16 + r + 8 * g;
    float v0 = acc0[r] + bA;
    float v1 = acc1[r] + bB;
    if (RELU) { v0 = fmaxf(v0, 0.f); v1 = fmaxf(v1, 0.f); }
    const long i0 = (long)row * ldc + col0;
    if (acc) { C[i0] += v0; C[i0 + 16] += v1; }
    else     { C[i0]  = v0; C[i0 + 16]  = v1; }
  }
}

// ============================================================================
// Flash attention: grid (S/64, B*H), block 128 (4 waves).
// Each wave owns 16 q rows; loop over 32 k-tiles of 64 with online softmax.
// qkv layout: [(s*B+b)*1536 + {0,512,1024} + h*64 + dh]
// ============================================================================
#define AST 72   // LDS row stride (halves), padded

__global__ __launch_bounds__(128)
void flash_attn(const float* __restrict__ qkv, float* __restrict__ ctx)
{
  __shared__ _Float16 Ql[64 * AST];
  __shared__ _Float16 Kl[64 * AST];
  __shared__ _Float16 Vl[64 * AST];
  __shared__ _Float16 Pl[64 * AST];
  const int qt = blockIdx.x;
  const int b  = blockIdx.y >> 3;
  const int h  = blockIdx.y & 7;
  const int t = threadIdx.x, wave = t >> 5, lane = t & 31;
  const int g = lane >> 4, nn = lane & 15;

  // stage Q (pre-scaled by 1/sqrt(hd) = 0.125)
  for (int idx = t; idx < 64 * 64; idx += 128) {
    int i = idx >> 6, dh = idx & 63;
    long row = (long)(qt * 64 + i) * B_ + b;
    Ql[i * AST + dh] = (_Float16)(qkv[row * 1536 + h * 64 + dh] * 0.125f);
  }
  __syncthreads();
  v16h aq0 = frag_a(&Ql[wave * 16 * AST], AST);        // dh 0..31
  v16h aq1 = frag_a(&Ql[wave * 16 * AST + 32], AST);   // dh 32..63

  float mrow[8], lrow[8];
  v8f o[4];
#pragma unroll
  for (int r = 0; r < 8; ++r) { mrow[r] = -3.0e38f; lrow[r] = 0.f; }
#pragma unroll
  for (int nt = 0; nt < 4; ++nt) o[nt] = v8fz();

  for (int kt = 0; kt < 32; ++kt) {
    for (int idx = t; idx < 64 * 64; idx += 128) {
      int i = idx >> 6, dh = idx & 63;
      long base = ((long)(kt * 64 + i) * B_ + b) * 1536 + h * 64 + dh;
      Kl[i * AST + dh] = (_Float16)qkv[base + 512];
      Vl[i * AST + dh] = (_Float16)qkv[base + 1024];
    }
    __syncthreads();

    // scores 16x64: S = Q * K^T   (B element (k,n) = Kl[n][k])
    v8f sc[4];
#pragma unroll
    for (int nt = 0; nt < 4; ++nt) {
      v16h bk0 = frag_bt(&Kl[nt * 16 * AST], AST);
      v16h bk1 = frag_bt(&Kl[nt * 16 * AST + 32], AST);
      sc[nt] = __builtin_amdgcn_wmma_f32_16x16x32_f16(false, aq0, false, bk0,
                                                      (short)0, v8fz(), false, false);
      sc[nt] = __builtin_amdgcn_wmma_f32_16x16x32_f16(false, aq1, false, bk1,
                                                      (short)0, sc[nt], false, false);
    }

    // online softmax; row (r+8g) lives on the 16 lanes of group g
#pragma unroll
    for (int r = 0; r < 8; ++r) {
      float tmax = sc[0][r];
#pragma unroll
      for (int nt = 1; nt < 4; ++nt) tmax = fmaxf(tmax, sc[nt][r]);
#pragma unroll
      for (int off = 8; off > 0; off >>= 1)
        tmax = fmaxf(tmax, __shfl_xor(tmax, off, 32));
      const float mn = fmaxf(mrow[r], tmax);
      const float corr = __expf(mrow[r] - mn);
      mrow[r] = mn;
      float ps = 0.f;
#pragma unroll
      for (int nt = 0; nt < 4; ++nt) {
        float p = __expf(sc[nt][r] - mn);
        sc[nt][r] = p;
        ps += p;
      }
#pragma unroll
      for (int off = 8; off > 0; off >>= 1) ps += __shfl_xor(ps, off, 32);
      lrow[r] = lrow[r] * corr + ps;
#pragma unroll
      for (int nt = 0; nt < 4; ++nt) o[nt][r] *= corr;
    }

    // stage P (C layout -> A layout via LDS)
#pragma unroll
    for (int r = 0; r < 8; ++r)
#pragma unroll
      for (int nt = 0; nt < 4; ++nt)
        Pl[(wave * 16 + r + 8 * g) * AST + nt * 16 + nn] = (_Float16)sc[nt][r];
    __syncthreads();
    v16h ap0 = frag_a(&Pl[wave * 16 * AST], AST);
    v16h ap1 = frag_a(&Pl[wave * 16 * AST + 32], AST);
#pragma unroll
    for (int nt = 0; nt < 4; ++nt) {
      v16h bv0 = frag_b(&Vl[nt * 16], AST);
      v16h bv1 = frag_b(&Vl[32 * AST + nt * 16], AST);
      o[nt] = __builtin_amdgcn_wmma_f32_16x16x32_f16(false, ap0, false, bv0,
                                                     (short)0, o[nt], false, false);
      o[nt] = __builtin_amdgcn_wmma_f32_16x16x32_f16(false, ap1, false, bv1,
                                                     (short)0, o[nt], false, false);
    }
    __syncthreads();
  }

#pragma unroll
  for (int r = 0; r < 8; ++r) {
    const float inv = 1.0f / lrow[r];
    long srow = (long)(qt * 64 + wave * 16 + r + 8 * g) * B_ + b;
    long base = srow * 512 + h * 64 + nn;
#pragma unroll
    for (int nt = 0; nt < 4; ++nt) ctx[base + nt * 16] = o[nt][r] * inv;
  }
}

// ============================================================================
// out = LayerNorm(a + res) * gamma + beta ; one wave per D=512 row
// ============================================================================
__global__ __launch_bounds__(256)
void add_layernorm(const float* __restrict__ a, const float* __restrict__ res,
                   const float* __restrict__ gam, const float* __restrict__ bet,
                   float* __restrict__ out)
{
  const int row = blockIdx.x * 8 + (threadIdx.x >> 5);
  const int lane = threadIdx.x & 31;
  const long base = (long)row * 512;
  float v[16];
  float s = 0.f;
#pragma unroll
  for (int i = 0; i < 16; ++i) {
    int c = lane + i * 32;
    v[i] = a[base + c] + res[base + c];
    s += v[i];
  }
#pragma unroll
  for (int off = 16; off > 0; off >>= 1) s += __shfl_xor(s, off, 32);
  const float mu = s * (1.f / 512.f);
  float var = 0.f;
#pragma unroll
  for (int i = 0; i < 16; ++i) { float d = v[i] - mu; var += d * d; }
#pragma unroll
  for (int off = 16; off > 0; off >>= 1) var += __shfl_xor(var, off, 32);
  const float inv = rsqrtf(var * (1.f / 512.f) + 1e-5f);
#pragma unroll
  for (int i = 0; i < 16; ++i) {
    int c = lane + i * 32;
    out[base + c] = (v[i] - mu) * inv * gam[c] + bet[c];
  }
}

// ============================================================================
// Routing: softmax over E=64 logits, argmax (lowest index on tie), gval.
// One wave per token; lane covers experts {lane, lane+32}.
// ============================================================================
__global__ __launch_bounds__(256)
void route(const float* __restrict__ logits, int* __restrict__ eidx,
           float* __restrict__ gval)
{
  const int tok = blockIdx.x * 8 + (threadIdx.x >> 5);
  const int lane = threadIdx.x & 31;
  const float l0 = logits[(long)tok * 64 + lane];
  const float l1 = logits[(long)tok * 64 + lane + 32];
  float bv; int bi;
  if (l0 >= l1) { bv = l0; bi = lane; } else { bv = l1; bi = lane + 32; }
#pragma unroll
  for (int off = 16; off > 0; off >>= 1) {
    float ov = __shfl_xor(bv, off, 32);
    int   oi = __shfl_xor(bi, off, 32);
    if (ov > bv || (ov == bv && oi < bi)) { bv = ov; bi = oi; }
  }
  float s = __expf(l0 - bv) + __expf(l1 - bv);
#pragma unroll
  for (int off = 16; off > 0; off >>= 1) s += __shfl_xor(s, off, 32);
  if (lane == 0) { eidx[tok] = bi; gval[tok] = 1.0f / s; }  // exp(0)/sum
}

// Deterministic cumulative position within each expert (token order).
__global__ void pos_scan(const int* __restrict__ eidx, int* __restrict__ pos)
{
  const int e = threadIdx.x;  // 64 threads, one per expert
  int cnt = 0;
  for (int n = 0; n < N_; ++n)
    if (eidx[n] == e) pos[n] = cnt++;
}

__global__ __launch_bounds__(128)
void scatter_tokens(const float* __restrict__ x, const int* __restrict__ eidx,
                    const int* __restrict__ pos, float* __restrict__ xe)
{
  const int n = blockIdx.x;
  const int p = pos[n];
  if (p >= CAP_) return;   // dropped (over capacity)
  const int e = eidx[n];
  const float* s = x + (long)n * 512;
  float* d = xe + ((long)e * CAP_ + p) * 512;
  for (int c = threadIdx.x; c < 512; c += 128) d[c] = s[c];
}

__global__ __launch_bounds__(128)
void gather_tokens(const float* __restrict__ ye, const float* __restrict__ b2,
                   const int* __restrict__ eidx, const int* __restrict__ pos,
                   const float* __restrict__ gval, float* __restrict__ moe)
{
  const int n = blockIdx.x;
  const int p = pos[n];
  const int e = eidx[n];
  float* d = moe + (long)n * 512;
  if (p >= CAP_) {   // over-capacity tokens contribute zero
    for (int c = threadIdx.x; c < 512; c += 128) d[c] = 0.f;
    return;
  }
  const float gv = gval[n];
  const float* s  = ye + ((long)e * CAP_ + p) * 512;
  const float* bb = b2 + (long)e * 512;
  for (int c = threadIdx.x; c < 512; c += 128) d[c] = (s[c] + bb[c]) * gv;
}

// ============================================================================
extern "C" void kernel_launch(void* const* d_in, const int* in_sizes, int n_in,
                              void* d_out, int out_size, void* d_ws, size_t ws_size,
                              hipStream_t stream)
{
  (void)in_sizes; (void)n_in; (void)out_size; (void)ws_size;
  const float* src   = (const float*)d_in[0];
  const float* inW   = (const float*)d_in[1];
  const float* inB   = (const float*)d_in[2];
  const float* outW  = (const float*)d_in[3];
  const float* outB  = (const float*)d_in[4];
  const float* ln1g  = (const float*)d_in[5];
  const float* ln1b  = (const float*)d_in[6];
  const float* gateW = (const float*)d_in[7];
  const float* w1    = (const float*)d_in[8];
  const float* b1    = (const float*)d_in[9];
  const float* w2    = (const float*)d_in[10];
  const float* b2    = (const float*)d_in[11];
  const float* ln2g  = (const float*)d_in[12];
  const float* ln2b  = (const float*)d_in[13];
  float* out = (float*)d_out;

  // Workspace layout (~374 MB): the N*1536 qkv region is reused for
  // attn_out and moe_out once attention has consumed qkv.
  float* ws       = (float*)d_ws;
  float* qkv      = ws;                          // N*1536
  float* attn_out = qkv;                         // alias (N*512), qkv dead
  float* moe      = qkv + (long)N_ * 512;        // alias (N*512), qkv dead
  float* ctx      = ws + (long)N_ * 1536;        // N*512
  float* x        = ctx + (long)N_ * 512;        // N*512
  float* logits   = x + (long)N_ * 512;          // N*64
  int*   eidx     = (int*)(logits + (long)N_ * 64);
  int*   pos      = eidx + N_;
  float* gval     = (float*)(pos + N_);
  float* xe       = gval + N_;                   // E*CAP*512
  float* hbuf     = xe + (long)E_ * CAP_ * 512;  // E*CAP*512 (F chunk)
  float* ye       = hbuf + (long)E_ * CAP_ * 512;// E*CAP*512

  // 1) qkv = src @ in_proj_w^T + b        (16384 x 1536, K=512)
  gemm_f16_wmma<true, false><<<dim3(1536 / 64, N_ / 64, 1), 256, 0, stream>>>(
      src, 512, 0, inW, 512, 0, inB, 0, qkv, 1536, 0, 512, 0);

  // 2) flash attention -> ctx
  flash_attn<<<dim3(S_ / 64, B_ * H_), 128, 0, stream>>>(qkv, ctx);

  // 3) attn_out = ctx @ out_proj_w^T + b
  gemm_f16_wmma<true, false><<<dim3(512 / 64, N_ / 64, 1), 256, 0, stream>>>(
      ctx, 512, 0, outW, 512, 0, outB, 0, attn_out, 512, 0, 512, 0);

  // 4) x = LN(src + attn_out)
  add_layernorm<<<N_ / 8, 256, 0, stream>>>(src, attn_out, ln1g, ln1b, x);

  // 5) gate logits = x @ gate_w          (Nc = 64)
  gemm_f16_wmma<false, false><<<dim3(1, N_ / 64, 1), 256, 0, stream>>>(
      x, 512, 0, gateW, 64, 0, nullptr, 0, logits, 64, 0, 512, 0);

  // 6-8) routing, positions, capacity scatter
  route<<<N_ / 8, 256, 0, stream>>>(logits, eidx, gval);
  pos_scan<<<1, 64, 0, stream>>>(eidx, pos);
  scatter_tokens<<<N_, 128, 0, stream>>>(x, eidx, pos, xe);

  // 9) expert FFN, F chunked by 512 to bound workspace:
  //    h_c = relu(xe @ w1[:,:,fc:fc+512] + b1[:,fc:fc+512]); ye += h_c @ w2[:,fc:fc+512,:]
  for (int c = 0; c < 4; ++c) {
    const int fc = c * 512;
    gemm_f16_wmma<false, true><<<dim3(8, 8, E_), 256, 0, stream>>>(
        xe, 512, (long)CAP_ * 512,
        w1 + fc, 2048, (long)512 * 2048,
        b1 + fc, 2048,
        hbuf, 512, (long)CAP_ * 512, 512, 0);
    gemm_f16_wmma<false, false><<<dim3(8, 8, E_), 256, 0, stream>>>(
        hbuf, 512, (long)CAP_ * 512,
        w2 + (long)fc * 512, 512, (long)2048 * 512,
        nullptr, 0,
        ye, 512, (long)CAP_ * 512, 512, c > 0 ? 1 : 0);
  }

  // 10) gather (+b2, *gval) -> moe ; 11) out = LN(x + moe)
  gather_tokens<<<N_, 128, 0, stream>>>(ye, b2, eidx, pos, gval, moe);
  add_layernorm<<<N_ / 8, 256, 0, stream>>>(x, moe, ln2g, ln2b, out);
}
